// GATEncoder_26182120636598
// MI455X (gfx1250) — matrix operand order repaired and checked by reading, not compiled, not run on hardware
//
#include <hip/hip_runtime.h>
#include <cstdint>
#include <cstddef>

typedef __attribute__((ext_vector_type(16))) _Float16 v16h;
typedef __attribute__((ext_vector_type(8)))  float    v8f;
typedef unsigned int v4u __attribute__((ext_vector_type(4)));
typedef int v8i __attribute__((ext_vector_type(8)));
typedef int v4i __attribute__((ext_vector_type(4)));

#define N_NODES 20000
#define N_EDGES 320000
#define N_ETOT  340000            // + self loops
#define IN_D    768
#define HID     128
#define NHEADS  4
#define H1      512               // 4*128
#define OUT_D   256
#define NGRAPH  64

// Tensor Data Mover path: device pass only; entry point chosen at compile time.
#if defined(__AMDGCN__)
#define USE_TDM 1
#else
#define USE_TDM 0
#endif

__device__ __forceinline__ float lrelu(float x) { return x > 0.0f ? x : 0.2f * x; }

__device__ __forceinline__ void atomicMaxF(float* addr, float v) {
    // standard IEEE ordering trick: works for mixed signs given -inf init
    if (v >= 0.0f) atomicMax(reinterpret_cast<int*>(addr), __float_as_int(v));
    else           atomicMin(reinterpret_cast<unsigned int*>(addr), __float_as_uint(v));
}

// ---------------------------------------------------------------- utilities
__global__ void fill_f32_kernel(float* __restrict__ p, float v, int n) {
    for (int i = blockIdx.x * blockDim.x + threadIdx.x; i < n; i += gridDim.x * blockDim.x)
        p[i] = v;
}

__global__ void cvt_f32_to_f16_kernel(const float* __restrict__ in, _Float16* __restrict__ out, int n) {
    for (int i = blockIdx.x * blockDim.x + threadIdx.x; i < n; i += gridDim.x * blockDim.x)
        out[i] = (_Float16)in[i];
}

// ---------------------------------------------------------------- WMMA GEMM
// C[M,Nc] = A[M,K] (f16, row-major) * B[K,Nc] (f16, row-major), f32 accumulate.
// Block tile 128x64, BK=32. 256 threads = 8 waves; wave (wm 0..3, wn 0..1)
// computes a 32x32 patch as 2x2 v_wmma_f32_16x16x32_f16 tiles.
// TDM path: wave 0 drives double-buffered tensor_load_to_lds (TENSORcnt),
// overlapping the DMA of tile k+1 with WMMA on tile k.
#define BM 128
#define BN 64
#define BK 32
#define LDA 36    // halves per As row: 16 DWORDs + 2 DWORD TDM pad
#define LDBS 68   // halves per Bs row: 32 DWORDs + 2 DWORD TDM pad

union U4H8 { uint4 u; _Float16 h[8]; };

#if USE_TDM
__device__ __forceinline__ void wait_tensorcnt0() {
#if __has_builtin(__builtin_amdgcn_s_wait_tensorcnt)
    __builtin_amdgcn_s_wait_tensorcnt((short)0);
#else
    asm volatile("s_wait_tensorcnt 0x0" ::: "memory");
#endif
}
__device__ __forceinline__ void wait_tensorcnt2() {
#if __has_builtin(__builtin_amdgcn_s_wait_tensorcnt)
    __builtin_amdgcn_s_wait_tensorcnt((short)2);
#else
    asm volatile("s_wait_tensorcnt 0x2" ::: "memory");
#endif
}

// Build a 2D-tile D# per CDNA5 ISA 8.3/8.4 and issue TENSOR_LOAD_TO_LDS.
// dims/strides in elements (f16); pad codes: interval 0=2DW..7=256DW, amount 0=1DW..
__device__ __forceinline__ void tdm_load_2d_f16(
    unsigned lds_addr, const _Float16* gptr,
    unsigned tile_d0, unsigned tile_d1,
    unsigned tensor_d0, unsigned tensor_d1,
    unsigned long long stride0,
    unsigned pad_interval_code, unsigned pad_amount_code)
{
    const unsigned long long ga = (unsigned long long)(size_t)gptr;
    v4u g0;
    g0.x = 1u;                                                   // count=1, user mode
    g0.y = lds_addr;                                             // lds_addr[31:0]
    g0.z = (unsigned)ga;                                         // global_addr[31:0]
    g0.w = (unsigned)((ga >> 32) & 0x01ffffffULL) | (2u << 30);  // global_addr[56:32] | type=2
    const unsigned w0 = (1u << 16)                    // data_size = 2 bytes
                      | (1u << 20)                    // pad_enable
                      | (pad_interval_code << 22)
                      | (pad_amount_code << 25);
    const unsigned w1 = (tensor_d0 & 0xffffu) << 16;                       // td0[15:0]
    const unsigned w2 = (tensor_d0 >> 16) | ((tensor_d1 & 0xffffu) << 16); // td0[31:16] | td1[15:0]
    const unsigned w3 = (tensor_d1 >> 16) | (tile_d0 << 16);               // td1[31:16] | tile_dim0
    const unsigned w4 = tile_d1;                                           // tile_dim1, tile_dim2=0
    const unsigned w5 = (unsigned)(stride0 & 0xffffffffULL);               // dim0_stride[31:0]
    const unsigned w6 = (unsigned)((stride0 >> 32) & 0xffffULL);           // dim0_stride[47:32]
    const unsigned w7 = 0u;                                                // dim1_stride unused (2D)
    v8i g1 = { (int)w0, (int)w1, (int)w2, (int)w3, (int)w4, (int)w5, (int)w6, (int)w7 };
    v4i zz = { 0, 0, 0, 0 };

#if __has_builtin(__builtin_amdgcn_tensor_load_to_lds_d2)
    // dedicated 2-group (<=2D tensor) form
    __builtin_amdgcn_tensor_load_to_lds_d2(g0, g1, 0);
    (void)zz;
#elif __has_builtin(__builtin_amdgcn_tensor_load_to_lds)
  #if __clang_major__ < 23
    // ROCm 7.2 (clang-22) 5-arg form
    __builtin_amdgcn_tensor_load_to_lds(g0, g1, zz, zz, 0);
  #else
    // amdgpu-toolchain (clang-23) 6-arg form: extra int32x8, zero-filled
    v8i z8 = { 0, 0, 0, 0, 0, 0, 0, 0 };
    __builtin_amdgcn_tensor_load_to_lds(g0, g1, zz, zz, z8, 0);
  #endif
#else
    // last resort: raw encoding via assembler (bypasses clang builtin)
    asm volatile("tensor_load_to_lds %0, %1, %2, %3"
                 :
                 : "s"(g0), "s"(g1), "s"(zz), "s"(zz)
                 : "memory");
#endif
}
#endif  // USE_TDM

__global__ __launch_bounds__(256) void gemm_f16_wmma_kernel(
    const _Float16* __restrict__ A, const _Float16* __restrict__ B,
    float* __restrict__ C, int M, int Nc, int K)
{
    const int tid  = threadIdx.x;
    const int lane = tid & 31;
    const int wid  = tid >> 5;
    const int wm   = wid & 3;            // 0..3 : 32-row slice
    const int wn   = wid >> 2;           // 0..1 : 32-col slice
    const int half_id = lane >> 4;       // 0/1
    const int l16     = lane & 15;

    const int block_m = blockIdx.y * BM;
    const int block_n = blockIdx.x * BN;

    v8f zero = {};
    v8f acc[2][2];
    acc[0][0] = zero; acc[0][1] = zero; acc[1][0] = zero; acc[1][1] = zero;

#if USE_TDM
    // ---------------- TDM double-buffered pipeline ----------------
    __shared__ _Float16 As[2][BM * LDA];
    __shared__ _Float16 Bs[2][BK * LDBS];   // row-major: Bs[k][col]

    const unsigned ldsA[2] = { (unsigned)(size_t)&As[0][0], (unsigned)(size_t)&As[1][0] };
    const unsigned ldsB[2] = { (unsigned)(size_t)&Bs[0][0], (unsigned)(size_t)&Bs[1][0] };
    const bool issuer = (wid == 0);
    const int nk = K / BK;

    auto issue_tiles = [&](int buf, int k0) {
        // A tile: BM x BK halves; row pad 16DW->+2DW  => LDS stride LDA=36 halves
        tdm_load_2d_f16(ldsA[buf], A + (size_t)block_m * K + k0,
                        BK, BM, (unsigned)(K - k0), (unsigned)(M - block_m),
                        (unsigned long long)K, 3u, 1u);
        // B tile: BK x BN halves; row pad 32DW->+2DW  => LDS stride LDBS=68 halves
        tdm_load_2d_f16(ldsB[buf], B + (size_t)k0 * Nc + block_n,
                        BN, BK, (unsigned)(Nc - block_n), (unsigned)(K - k0),
                        (unsigned long long)Nc, 4u, 1u);
    };

    if (issuer) issue_tiles(0, 0);

    for (int kt = 0; kt < nk; ++kt) {
        const int cur = kt & 1;
        if (issuer) {
            if (kt + 1 < nk) {
                issue_tiles(cur ^ 1, (kt + 1) * BK);
                wait_tensorcnt2();  // current tile retired; next tile still in flight
            } else {
                wait_tensorcnt0();
            }
        }
        __syncthreads();

        v16h afrag[2], bfrag[2];
#pragma unroll
        for (int tm = 0; tm < 2; ++tm) {
            const _Float16* ap = &As[cur][(wm * 32 + tm * 16 + l16) * LDA];
#pragma unroll
            for (int i = 0; i < 8; ++i) {
                const int k = (i < 4) ? (half_id * 8 + 2 * i)
                                      : (16 + half_id * 8 + 2 * (i - 4));
                afrag[tm][2 * i]     = ap[k];
                afrag[tm][2 * i + 1] = ap[k + 1];
            }
        }
#pragma unroll
        for (int tn = 0; tn < 2; ++tn) {
            const int col = wn * 32 + tn * 16 + l16;
            const _Float16* bp = &Bs[cur][0];
#pragma unroll
            for (int i = 0; i < 8; ++i) {
                const int k = (i < 4) ? (half_id * 8 + 2 * i) : (16 + half_id * 8 + 2 * (i - 4));
                bfrag[tn][2 * i]     = bp[k * LDBS + col];
                bfrag[tn][2 * i + 1] = bp[(k + 1) * LDBS + col];
            }
        }

#pragma unroll
        for (int tm = 0; tm < 2; ++tm)
#pragma unroll
            for (int tn = 0; tn < 2; ++tn)
                acc[tm][tn] = __builtin_amdgcn_wmma_f32_16x16x32_f16(
                    false, afrag[tm], false, bfrag[tn],
                    (short)0, acc[tm][tn], false, false);

        __syncthreads();   // protect buf (cur^1 issued next iter) until all waves done
    }
#else
    // ---------------- manual-staging fallback (host pass / no TDM) ----------------
    __shared__ _Float16 As[BM * LDA];
    __shared__ _Float16 BsT[BN * LDA];   // B transposed: BsT[col][k]

    const int a_row = tid >> 1;
    const int a_seg = (tid & 1) * 16;
    const int b_k   = tid >> 3;
    const int b_seg = (tid & 7) * 8;

    for (int k0 = 0; k0 < K; k0 += BK) {
        {
            const int grow = block_m + a_row;
            uint2* ldst = reinterpret_cast<uint2*>(&As[a_row * LDA + a_seg]);
            if (grow < M) {
                const uint2* gsrc =
                    reinterpret_cast<const uint2*>(A + (size_t)grow * K + k0 + a_seg);
                ldst[0] = gsrc[0]; ldst[1] = gsrc[1]; ldst[2] = gsrc[2]; ldst[3] = gsrc[3];
                if (k0 + BK < K)
                    __builtin_prefetch(A + (size_t)grow * K + k0 + BK + a_seg, 0, 3);
            } else {
                uint2 z; z.x = 0u; z.y = 0u;
                ldst[0] = z; ldst[1] = z; ldst[2] = z; ldst[3] = z;
            }
        }
        {
            U4H8 t;
            t.u = *reinterpret_cast<const uint4*>(B + (size_t)(k0 + b_k) * Nc + block_n + b_seg);
#pragma unroll
            for (int j = 0; j < 8; ++j)
                BsT[(b_seg + j) * LDA + b_k] = t.h[j];
        }
        __syncthreads();

        v16h afrag[2], bfrag[2];
#pragma unroll
        for (int tm = 0; tm < 2; ++tm) {
            const _Float16* ap = &As[(wm * 32 + tm * 16 + l16) * LDA];
#pragma unroll
            for (int i = 0; i < 8; ++i) {
                const int k = (i < 4) ? (half_id * 8 + 2 * i)
                                      : (16 + half_id * 8 + 2 * (i - 4));
                afrag[tm][2 * i]     = ap[k];
                afrag[tm][2 * i + 1] = ap[k + 1];
            }
        }
#pragma unroll
        for (int tn = 0; tn < 2; ++tn) {
            const _Float16* bp = &BsT[(wn * 32 + tn * 16 + l16) * LDA];
#pragma unroll
            for (int i = 0; i < 8; ++i) {
                const int k = (i < 4) ? (half_id * 8 + 2 * i) : (16 + half_id * 8 + 2 * (i - 4));
                bfrag[tn][2 * i]     = bp[k];
                bfrag[tn][2 * i + 1] = bp[k + 1];
            }
        }

#pragma unroll
        for (int tm = 0; tm < 2; ++tm)
#pragma unroll
            for (int tn = 0; tn < 2; ++tn)
                acc[tm][tn] = __builtin_amdgcn_wmma_f32_16x16x32_f16(
                    false, afrag[tm], false, bfrag[tn],
                    (short)0, acc[tm][tn], false, false);

        __syncthreads();
    }
#endif

    // ---- epilogue: f32 C/D layout (VGPR r -> M = half_id*8 + r, N = l16) ----
#pragma unroll
    for (int tm = 0; tm < 2; ++tm) {
#pragma unroll
        for (int tn = 0; tn < 2; ++tn) {
            const int col = block_n + wn * 32 + tn * 16 + l16;
#pragma unroll
            for (int r = 0; r < 8; ++r) {
                const int row = block_m + wm * 32 + tm * 16 + half_id * 8 + r;
                if (row < M) C[(size_t)row * Nc + col] = acc[tm][tn][r];
            }
        }
    }
}

// ---------------------------------------------------------------- attention scores
// one wave per node: s_src[n,h] = dot(h[n, h*C:(h+1)*C], a_src[h]); same for dst
__global__ __launch_bounds__(256) void score_kernel(
    const float* __restrict__ h, const float* __restrict__ a_src,
    const float* __restrict__ a_dst, float* __restrict__ ssrc,
    float* __restrict__ sdst, int n, int heads, int C)
{
    const int node = blockIdx.x * (blockDim.x >> 5) + (threadIdx.x >> 5);
    const int lane = threadIdx.x & 31;
    if (node >= n) return;
    const int cols = heads * C;
    float ss[4] = {0.f, 0.f, 0.f, 0.f};
    float sd[4] = {0.f, 0.f, 0.f, 0.f};
    const float* hp = h + (size_t)node * cols;
    for (int c = lane; c < cols; c += 32) {
        const float v = hp[c];
        const int hd = c / C;
        ss[hd] += v * a_src[c];
        sd[hd] += v * a_dst[c];
    }
#pragma unroll
    for (int off = 16; off > 0; off >>= 1)
#pragma unroll
        for (int hd = 0; hd < 4; ++hd) {
            ss[hd] += __shfl_xor(ss[hd], off, 32);
            sd[hd] += __shfl_xor(sd[hd], off, 32);
        }
    if (lane == 0)
        for (int hd = 0; hd < heads; ++hd) {
            ssrc[node * heads + hd] = ss[hd];
            sdst[node * heads + hd] = sd[hd];
        }
}

// ---------------------------------------------------------------- edge kernels
__global__ void edge_max_kernel(const int* __restrict__ src, const int* __restrict__ dst,
                                const float* __restrict__ ssrc, const float* __restrict__ sdst,
                                float* __restrict__ smax, int heads)
{
    const int total = N_ETOT * heads;
    for (int idx = blockIdx.x * blockDim.x + threadIdx.x; idx < total;
         idx += gridDim.x * blockDim.x) {
        const int e = idx / heads, hd = idx - e * heads;
        const int s = (e < N_EDGES) ? src[e] : (e - N_EDGES);
        const int d = (e < N_EDGES) ? dst[e] : (e - N_EDGES);
        const float eh = lrelu(ssrc[s * heads + hd] + sdst[d * heads + hd]);
        atomicMaxF(&smax[d * heads + hd], eh);
    }
}

__global__ void edge_sum_kernel(const int* __restrict__ src, const int* __restrict__ dst,
                                const float* __restrict__ ssrc, const float* __restrict__ sdst,
                                const float* __restrict__ smax, float* __restrict__ sden, int heads)
{
    const int total = N_ETOT * heads;
    for (int idx = blockIdx.x * blockDim.x + threadIdx.x; idx < total;
         idx += gridDim.x * blockDim.x) {
        const int e = idx / heads, hd = idx - e * heads;
        const int s = (e < N_EDGES) ? src[e] : (e - N_EDGES);
        const int d = (e < N_EDGES) ? dst[e] : (e - N_EDGES);
        const float eh = lrelu(ssrc[s * heads + hd] + sdst[d * heads + hd]);
        atomicAdd(&sden[d * heads + hd], __expf(eh - smax[d * heads + hd]));
    }
}

// out[dst] += p * h[src]  (denominator folded into epilogue)
__global__ void edge_agg_kernel(const int* __restrict__ src, const int* __restrict__ dst,
                                const float* __restrict__ ssrc, const float* __restrict__ sdst,
                                const float* __restrict__ smax,
                                const float* __restrict__ hlin, float* __restrict__ hagg,
                                int heads, int C)
{
    const int cols  = heads * C;
    const int cols4 = cols >> 2;
    const int total = N_ETOT * cols4;   // 43.5M max, fits int
    for (int idx = blockIdx.x * blockDim.x + threadIdx.x; idx < total;
         idx += gridDim.x * blockDim.x) {
        const int e = idx / cols4;
        const int c0 = (idx - e * cols4) * 4;
        const int hd = c0 / C;
        const int s = (e < N_EDGES) ? src[e] : (e - N_EDGES);
        const int d = (e < N_EDGES) ? dst[e] : (e - N_EDGES);
        const float eh = lrelu(ssrc[s * heads + hd] + sdst[d * heads + hd]);
        const float p  = __expf(eh - smax[d * heads + hd]);
        const float4 v = *reinterpret_cast<const float4*>(&hlin[(size_t)s * cols + c0]);
        float* o = &hagg[(size_t)d * cols + c0];
        atomicAdd(o + 0, p * v.x);
        atomicAdd(o + 1, p * v.y);
        atomicAdd(o + 2, p * v.z);
        atomicAdd(o + 3, p * v.w);
    }
}

// hagg = hagg / (den + 1e-16) + bias   (in place)
__global__ void post_div_kernel(float* __restrict__ hagg, const float* __restrict__ sden,
                                const float* __restrict__ bias, int n, int heads, int C)
{
    const int cols = heads * C;
    const int total = n * cols;
    for (int idx = blockIdx.x * blockDim.x + threadIdx.x; idx < total;
         idx += gridDim.x * blockDim.x) {
        const int node = idx / cols;
        const int c = idx - node * cols;
        const int hd = c / C;
        hagg[idx] = hagg[idx] / (sden[node * heads + hd] + 1e-16f) + bias[c];
    }
}

// per-channel mean/var (one block per channel)
__global__ __launch_bounds__(256) void bn_stats_kernel(const float* __restrict__ x,
                                                       float* __restrict__ stats, int n, int cols)
{
    __shared__ float s0[256], s1[256];
    const int ch = blockIdx.x;
    const int tid = threadIdx.x;
    float a = 0.f, b = 0.f;
    for (int r = tid; r < n; r += 256) {
        const float v = x[(size_t)r * cols + ch];
        a += v; b += v * v;
    }
    s0[tid] = a; s1[tid] = b;
    __syncthreads();
    for (int st = 128; st > 0; st >>= 1) {
        if (tid < st) { s0[tid] += s0[tid + st]; s1[tid] += s1[tid + st]; }
        __syncthreads();
    }
    if (tid == 0) {
        const float mu = s0[0] / (float)n;
        stats[ch] = mu;
        stats[cols + ch] = s1[0] / (float)n - mu * mu;
    }
}

// y = lrelu(gamma*(x-mu)*rsqrt(var+eps)+beta) -> f16 (next GEMM operand)
__global__ void bn_apply_kernel(const float* __restrict__ x, const float* __restrict__ stats,
                                const float* __restrict__ gamma, const float* __restrict__ beta,
                                _Float16* __restrict__ out, int n, int cols)
{
    const int total = n * cols;
    for (int idx = blockIdx.x * blockDim.x + threadIdx.x; idx < total;
         idx += gridDim.x * blockDim.x) {
        const int c = idx % cols;
        const float mu = stats[c];
        const float var = stats[cols + c];
        float y = gamma[c] * (x[idx] - mu) * rsqrtf(var + 1e-5f) + beta[c];
        out[idx] = (_Float16)lrelu(y);
    }
}

// layer-3 epilogue fused with graph mean-pool accumulation
__global__ void post3_pool_kernel(const float* __restrict__ hagg, const float* __restrict__ sden,
                                  const float* __restrict__ b3, const int* __restrict__ batch,
                                  float* __restrict__ outsum, int n)
{
    const int total = n * OUT_D;
    for (int idx = blockIdx.x * blockDim.x + threadIdx.x; idx < total;
         idx += gridDim.x * blockDim.x) {
        const int node = idx / OUT_D;
        const int c = idx - node * OUT_D;
        const float v = lrelu(hagg[idx] / (sden[node] + 1e-16f) + b3[c]);
        atomicAdd(&outsum[(size_t)batch[node] * OUT_D + c], v);
    }
}

__global__ void count_kernel(const int* __restrict__ batch, float* __restrict__ cnt, int n) {
    for (int i = blockIdx.x * blockDim.x + threadIdx.x; i < n; i += gridDim.x * blockDim.x)
        atomicAdd(&cnt[batch[i]], 1.0f);
}

__global__ void finalize_kernel(float* __restrict__ out, const float* __restrict__ cnt) {
    const int idx = blockIdx.x * blockDim.x + threadIdx.x;
    if (idx < NGRAPH * OUT_D)
        out[idx] = out[idx] / fmaxf(cnt[idx / OUT_D], 1.0f);
}

// ---------------------------------------------------------------- host driver
extern "C" void kernel_launch(void* const* d_in, const int* in_sizes, int n_in,
                              void* d_out, int out_size, void* d_ws, size_t ws_size,
                              hipStream_t stream) {
    (void)in_sizes; (void)n_in; (void)out_size; (void)ws_size;

    const float* x      = (const float*)d_in[0];
    const int*   eidx   = (const int*)d_in[1];
    const int*   batch  = (const int*)d_in[2];
    const float* W1     = (const float*)d_in[3];
    const float* a_src1 = (const float*)d_in[4];
    const float* a_dst1 = (const float*)d_in[5];
    const float* b1     = (const float*)d_in[6];
    const float* gamma1 = (const float*)d_in[7];
    const float* beta1  = (const float*)d_in[8];
    const float* W2     = (const float*)d_in[9];
    const float* a_src2 = (const float*)d_in[10];
    const float* a_dst2 = (const float*)d_in[11];
    const float* b2     = (const float*)d_in[12];
    const float* gamma2 = (const float*)d_in[13];
    const float* beta2  = (const float*)d_in[14];
    const float* W3     = (const float*)d_in[15];
    const float* a_src3 = (const float*)d_in[16];
    const float* a_dst3 = (const float*)d_in[17];
    const float* b3     = (const float*)d_in[18];

    const int* esrc = eidx;
    const int* edst = eidx + N_EDGES;
    float* out = (float*)d_out;

    // workspace carve-up
    char* ws = (char*)d_ws;
    size_t off = 0;
    auto carve = [&](size_t bytes) {
        char* p = ws + off;
        off = (off + bytes + 255) & ~(size_t)255;
        return p;
    };
    _Float16* xf16  = (_Float16*)carve((size_t)N_NODES * IN_D * 2);
    _Float16* w1f16 = (_Float16*)carve((size_t)IN_D * H1 * 2);
    _Float16* w2f16 = (_Float16*)carve((size_t)H1 * H1 * 2);
    _Float16* w3f16 = (_Float16*)carve((size_t)H1 * OUT_D * 2);
    float*    hlin  = (float*)carve((size_t)N_NODES * H1 * 4);
    float*    hagg  = (float*)carve((size_t)N_NODES * H1 * 4);
    _Float16* hf16  = (_Float16*)carve((size_t)N_NODES * H1 * 2);
    float*    ssrc  = (float*)carve((size_t)N_NODES * NHEADS * 4);
    float*    sdst  = (float*)carve((size_t)N_NODES * NHEADS * 4);
    float*    smax  = (float*)carve((size_t)N_NODES * NHEADS * 4);
    float*    sden  = (float*)carve((size_t)N_NODES * NHEADS * 4);
    float*    bnst  = (float*)carve((size_t)2 * H1 * 4);
    float*    cnt   = (float*)carve((size_t)NGRAPH * 4);

    const int TPB = 256;
    const int GS  = 4096;   // grid-stride block count for big elementwise/edge kernels
    const float NEG_INF = -__builtin_huge_valf();

    // ---- f32 -> f16 operand conversion ----
    cvt_f32_to_f16_kernel<<<GS, TPB, 0, stream>>>(x,  xf16,  N_NODES * IN_D);
    cvt_f32_to_f16_kernel<<<GS, TPB, 0, stream>>>(W1, w1f16, IN_D * H1);
    cvt_f32_to_f16_kernel<<<GS, TPB, 0, stream>>>(W2, w2f16, H1 * H1);
    cvt_f32_to_f16_kernel<<<GS, TPB, 0, stream>>>(W3, w3f16, H1 * OUT_D);

    const dim3 gemm_grid12(H1 / BN, (N_NODES + BM - 1) / BM);
    const dim3 gemm_grid3(OUT_D / BN, (N_NODES + BM - 1) / BM);
    const int score_blocks = (N_NODES + 7) / 8;

    // ================= layer 1 =================
    gemm_f16_wmma_kernel<<<gemm_grid12, TPB, 0, stream>>>(xf16, w1f16, hlin, N_NODES, H1, IN_D);
    score_kernel<<<score_blocks, TPB, 0, stream>>>(hlin, a_src1, a_dst1, ssrc, sdst, N_NODES, NHEADS, HID);
    fill_f32_kernel<<<GS, TPB, 0, stream>>>(smax, NEG_INF, N_NODES * NHEADS);
    fill_f32_kernel<<<GS, TPB, 0, stream>>>(sden, 0.0f, N_NODES * NHEADS);
    fill_f32_kernel<<<GS, TPB, 0, stream>>>(hagg, 0.0f, N_NODES * H1);
    edge_max_kernel<<<GS, TPB, 0, stream>>>(esrc, edst, ssrc, sdst, smax, NHEADS);
    edge_sum_kernel<<<GS, TPB, 0, stream>>>(esrc, edst, ssrc, sdst, smax, sden, NHEADS);
    edge_agg_kernel<<<GS, TPB, 0, stream>>>(esrc, edst, ssrc, sdst, smax, hlin, hagg, NHEADS, HID);
    post_div_kernel<<<GS, TPB, 0, stream>>>(hagg, sden, b1, N_NODES, NHEADS, HID);
    bn_stats_kernel<<<H1, TPB, 0, stream>>>(hagg, bnst, N_NODES, H1);
    bn_apply_kernel<<<GS, TPB, 0, stream>>>(hagg, bnst, gamma1, beta1, hf16, N_NODES, H1);

    // ================= layer 2 =================
    gemm_f16_wmma_kernel<<<gemm_grid12, TPB, 0, stream>>>(hf16, w2f16, hlin, N_NODES, H1, H1);
    score_kernel<<<score_blocks, TPB, 0, stream>>>(hlin, a_src2, a_dst2, ssrc, sdst, N_NODES, NHEADS, HID);
    fill_f32_kernel<<<GS, TPB, 0, stream>>>(smax, NEG_INF, N_NODES * NHEADS);
    fill_f32_kernel<<<GS, TPB, 0, stream>>>(sden, 0.0f, N_NODES * NHEADS);
    fill_f32_kernel<<<GS, TPB, 0, stream>>>(hagg, 0.0f, N_NODES * H1);
    edge_max_kernel<<<GS, TPB, 0, stream>>>(esrc, edst, ssrc, sdst, smax, NHEADS);
    edge_sum_kernel<<<GS, TPB, 0, stream>>>(esrc, edst, ssrc, sdst, smax, sden, NHEADS);
    edge_agg_kernel<<<GS, TPB, 0, stream>>>(esrc, edst, ssrc, sdst, smax, hlin, hagg, NHEADS, HID);
    post_div_kernel<<<GS, TPB, 0, stream>>>(hagg, sden, b2, N_NODES, NHEADS, HID);
    bn_stats_kernel<<<H1, TPB, 0, stream>>>(hagg, bnst, N_NODES, H1);
    bn_apply_kernel<<<GS, TPB, 0, stream>>>(hagg, bnst, gamma2, beta2, hf16, N_NODES, H1);

    // ================= layer 3 (1 head, 256 ch) + mean pool =================
    gemm_f16_wmma_kernel<<<gemm_grid3, TPB, 0, stream>>>(hf16, w3f16, hlin, N_NODES, OUT_D, H1);
    score_kernel<<<score_blocks, TPB, 0, stream>>>(hlin, a_src3, a_dst3, ssrc, sdst, N_NODES, 1, OUT_D);
    fill_f32_kernel<<<GS, TPB, 0, stream>>>(smax, NEG_INF, N_NODES);
    fill_f32_kernel<<<GS, TPB, 0, stream>>>(sden, 0.0f, N_NODES);
    fill_f32_kernel<<<GS, TPB, 0, stream>>>(hagg, 0.0f, N_NODES * OUT_D);
    edge_max_kernel<<<GS, TPB, 0, stream>>>(esrc, edst, ssrc, sdst, smax, 1);
    edge_sum_kernel<<<GS, TPB, 0, stream>>>(esrc, edst, ssrc, sdst, smax, sden, 1);
    edge_agg_kernel<<<GS, TPB, 0, stream>>>(esrc, edst, ssrc, sdst, smax, hlin, hagg, 1, OUT_D);
    fill_f32_kernel<<<64, TPB, 0, stream>>>(out, 0.0f, NGRAPH * OUT_D);
    fill_f32_kernel<<<1, TPB, 0, stream>>>(cnt, 0.0f, NGRAPH);
    post3_pool_kernel<<<GS, TPB, 0, stream>>>(hagg, sden, b3, batch, out, N_NODES);
    count_kernel<<<GS, TPB, 0, stream>>>(batch, cnt, N_NODES);
    finalize_kernel<<<(NGRAPH * OUT_D + TPB - 1) / TPB, TPB, 0, stream>>>(out, cnt);
}
